// Sru_36120674959532
// MI455X (gfx1250) — compile-verified, matrix operand order checked by reading, not run on hardware
//
#include <hip/hip_runtime.h>
#include <hip/hip_bf16.h>
#include <stdint.h>

typedef __bf16   bf16x16 __attribute__((ext_vector_type(16)));
typedef float    v8f     __attribute__((ext_vector_type(8)));
typedef uint32_t u32x4   __attribute__((ext_vector_type(4)));
typedef unsigned short ushort_t;

#define D_IN    1024
#define H_DIM   512
#define N3H     1536
#define NC      15

// ---------- address-space(1) helpers: force global_load/global_store ----------
template <typename T>
__device__ __forceinline__ T gld(const void* p) {
    return *(const __attribute__((address_space(1))) T*)(uintptr_t)p;
}
template <typename T>
__device__ __forceinline__ void gst(void* p, T v) {
    *(__attribute__((address_space(1))) T*)(uintptr_t)p = v;
}

// ---------- helpers ----------
__device__ __forceinline__ ushort_t f2bf(float f) {
    uint32_t x = __float_as_uint(f);
    uint32_t r = x + 0x7FFFu + ((x >> 16) & 1u);   // round-to-nearest-even
    return (ushort_t)(r >> 16);
}

union Frag {
    bf16x16 v;
    u32x4   q[2];
};

// ---------- convert f32 -> bf16 (elementwise) ----------
__global__ void cvt_f32_bf16(const float* __restrict__ in, ushort_t* __restrict__ out, int n) {
    int i = blockIdx.x * blockDim.x + threadIdx.x;
    if (i < n) gst<ushort_t>(out + i, f2bf(gld<float>(in + i)));
}

// ---------- convert + transpose weights: W [K,N] f32 -> Wt [N,K] bf16 ----------
__global__ void cvt_transpose(const float* __restrict__ W, ushort_t* __restrict__ Wt, int K, int N) {
    int i = blockIdx.x * blockDim.x + threadIdx.x;
    if (i < K * N) {
        int n = i / K, k = i - n * K;
        gst<ushort_t>(Wt + i, f2bf(gld<float>(W + (size_t)k * N + n)));
    }
}

// ---------- WMMA GEMM: C[M,N] f32 = A[M,K] bf16 @ Bt[N,K]^T bf16 ----------
// block = 256 threads (8 waves), wave tile 64x32, block tile 128x128.
// blockIdx.z selects (B0,C0) or (B1,C1) so fwd/bwd GEMMs run in one launch.
__global__ void __launch_bounds__(256)
wmma_gemm_bf16(const ushort_t* __restrict__ A,
               const ushort_t* __restrict__ B0, const ushort_t* __restrict__ B1,
               float* __restrict__ C0, float* __restrict__ C1,
               int M, int N, int K) {
    const ushort_t* B = (blockIdx.z == 0) ? B0 : B1;
    float*          C = (blockIdx.z == 0) ? C0 : C1;

    const int lane = threadIdx.x & 31;
    const int wave = threadIdx.x >> 5;
    const int wm   = wave & 1;          // 2 waves along M
    const int wn   = wave >> 1;         // 4 waves along N
    const int mWave = blockIdx.y * 128 + wm * 64;
    const int nWave = blockIdx.x * 128 + wn * 32;
    const int half  = lane >> 4;        // 0: K 0..7/16..23, 1: K 8..15/24..31
    const int mr    = lane & 15;

    v8f acc[4][2];
    #pragma unroll
    for (int i = 0; i < 4; ++i)
        #pragma unroll
        for (int j = 0; j < 2; ++j)
            acc[i][j] = (v8f)(0.0f);

    const ushort_t* aptr[4];
    const ushort_t* bptr[2];
    #pragma unroll
    for (int i = 0; i < 4; ++i)
        aptr[i] = A + (size_t)(mWave + i * 16 + mr) * K + half * 8;
    #pragma unroll
    for (int j = 0; j < 2; ++j)
        bptr[j] = B + (size_t)(nWave + j * 16 + mr) * K + half * 8;

    for (int k0 = 0; k0 < K; k0 += 32) {
        Frag af[4], bfr[2];
        #pragma unroll
        for (int i = 0; i < 4; ++i) {
            af[i].q[0] = gld<u32x4>(aptr[i]);
            af[i].q[1] = gld<u32x4>(aptr[i] + 16);
            aptr[i] += 32;
        }
        #pragma unroll
        for (int j = 0; j < 2; ++j) {
            bfr[j].q[0] = gld<u32x4>(bptr[j]);
            bfr[j].q[1] = gld<u32x4>(bptr[j] + 16);
            bptr[j] += 32;
        }
        #pragma unroll
        for (int i = 0; i < 4; ++i)
            #pragma unroll
            for (int j = 0; j < 2; ++j)
                acc[i][j] = __builtin_amdgcn_wmma_f32_16x16x32_bf16(
                    false, af[i].v, false, bfr[j].v, (short)0, acc[i][j], false, false);
    }

    // D layout: lanes 0-15 -> N=lane, M=r; lanes 16-31 -> N=lane-16, M=8+r
    #pragma unroll
    for (int i = 0; i < 4; ++i)
        #pragma unroll
        for (int j = 0; j < 2; ++j) {
            float* cp = C + (size_t)(mWave + i * 16 + half * 8) * N + (nWave + j * 16 + mr);
            #pragma unroll
            for (int r = 0; r < 8; ++r)
                gst<float>(cp + (size_t)r * N, acc[i][j][r]);
        }
}

// ---------- SRU scan: one block per direction, thread = channel ----------
// blockIdx.x == 0 : forward (U=Uf, highway cols 0..H-1, out cols 0..H-1)
// blockIdx.x == 1 : backward (U=Ub, highway cols H..2H-1, out cols H..2H-1)
__global__ void __launch_bounds__(512)
sru_scan(const float* __restrict__ Uf, const float* __restrict__ Ub,
         const float* __restrict__ vF, const float* __restrict__ bF,
         const float* __restrict__ vB, const float* __restrict__ bB,
         const float* __restrict__ xres,   // [T, 2H] layer input
         float* __restrict__ out,          // [T, 2H]
         int T) {
    const int h   = threadIdx.x;
    const bool bwd = (blockIdx.x == 1);
    const float* U = bwd ? Ub : Uf;
    const float* v = bwd ? vB : vF;
    const float* bb = bwd ? bB : bF;
    const int col  = bwd ? (H_DIM + h) : h;

    const float vf  = gld<float>(v + h);
    const float vr  = gld<float>(v + H_DIM + h);
    const float b_f = gld<float>(bb + h);
    const float b_r = gld<float>(bb + H_DIM + h);

    float c = 0.0f;
    for (int i = 0; i < T; ++i) {
        const int t = bwd ? (T - 1 - i) : i;
        const size_t ub = (size_t)t * N3H;
        // prefetch a few timesteps ahead into cache
        if (i + 4 < T) {
            const int tn = bwd ? (T - 1 - (i + 4)) : (i + 4);
            __builtin_prefetch(&U[(size_t)tn * N3H + h], 0, 1);
            __builtin_prefetch(&xres[(size_t)tn * (2 * H_DIM) + col], 0, 1);
        }
        const float u0 = gld<float>(U + ub + h);
        const float u1 = gld<float>(U + ub + H_DIM + h);
        const float u2 = gld<float>(U + ub + 2 * H_DIM + h);
        const float xr = gld<float>(xres + (size_t)t * (2 * H_DIM) + col);

        const float g1 = 1.0f / (1.0f + __expf(-(u1 + vf * c + b_f)));
        c = g1 * c + (1.0f - g1) * u0;
        const float g2 = 1.0f / (1.0f + __expf(-(u2 + vr * c + b_r)));
        gst<float>(out + (size_t)t * (2 * H_DIM) + col, g2 * c + (1.0f - g2) * xr);
    }
}

// ---------- final FC: one wave per row t, lanes split K, shuffle-reduce ----------
__global__ void __launch_bounds__(256)
fc_kernel(const float* __restrict__ h, const float* __restrict__ W,
          const float* __restrict__ bias, float* __restrict__ out, int T) {
    const int K = 2 * H_DIM;
    const int lane = threadIdx.x & 31;
    const int wave = threadIdx.x >> 5;
    const int t = blockIdx.x * (blockDim.x >> 5) + wave;
    if (t >= T) return;

    const float* hr = h + (size_t)t * K;
    float acc[NC];
    #pragma unroll
    for (int c = 0; c < NC; ++c) acc[c] = 0.0f;

    for (int k = lane; k < K; k += 32) {
        const float hv = gld<float>(hr + k);
        const float* wr = W + (size_t)k * NC;
        #pragma unroll
        for (int c = 0; c < NC; ++c) acc[c] += hv * gld<float>(wr + c);
    }
    #pragma unroll
    for (int off = 16; off > 0; off >>= 1)
        #pragma unroll
        for (int c = 0; c < NC; ++c) acc[c] += __shfl_down(acc[c], off, 32);

    if (lane == 0) {
        float* o = out + (size_t)t * NC;
        #pragma unroll
        for (int c = 0; c < NC; ++c) gst<float>(o + c, acc[c] + gld<float>(bias + c));
    }
}

// ---------- launch ----------
extern "C" void kernel_launch(void* const* d_in, const int* in_sizes, int n_in,
                              void* d_out, int out_size, void* d_ws, size_t ws_size,
                              hipStream_t stream) {
    const float* x    = (const float*)d_in[0];
    const float* w0f  = (const float*)d_in[1];
    const float* v0f  = (const float*)d_in[2];
    const float* b0f  = (const float*)d_in[3];
    const float* w0b  = (const float*)d_in[4];
    const float* v0b  = (const float*)d_in[5];
    const float* b0b  = (const float*)d_in[6];
    const float* w1f  = (const float*)d_in[7];
    const float* v1f  = (const float*)d_in[8];
    const float* b1f  = (const float*)d_in[9];
    const float* w1b  = (const float*)d_in[10];
    const float* v1b  = (const float*)d_in[11];
    const float* b1b  = (const float*)d_in[12];
    const float* fc_w = (const float*)d_in[13];
    const float* fc_b = (const float*)d_in[14];
    float* outp = (float*)d_out;

    const int T = in_sizes[0] / D_IN;   // 16384

    // --- carve workspace ---
    size_t off = 0;
    auto carve = [&](size_t bytes) -> void* {
        void* p = (char*)d_ws + off;
        off += (bytes + 255) & ~(size_t)255;
        return p;
    };
    ushort_t* x_bf  = (ushort_t*)carve((size_t)T * D_IN * 2);       // 32 MB
    ushort_t* wt0f  = (ushort_t*)carve((size_t)N3H * D_IN * 2);     // 3 MB
    ushort_t* wt0b  = (ushort_t*)carve((size_t)N3H * D_IN * 2);
    ushort_t* wt1f  = (ushort_t*)carve((size_t)N3H * D_IN * 2);
    ushort_t* wt1b  = (ushort_t*)carve((size_t)N3H * D_IN * 2);
    float*    Uf    = (float*)carve((size_t)T * N3H * 4);           // ~101 MB
    float*    Ub    = (float*)carve((size_t)T * N3H * 4);           // ~101 MB
    float*    h1    = (float*)carve((size_t)T * D_IN * 4);          // 64 MB
    ushort_t* h1bf  = (ushort_t*)carve((size_t)T * D_IN * 2);       // 32 MB
    float*    h2    = (float*)carve((size_t)T * D_IN * 4);          // 64 MB

    const int nX = T * D_IN;
    const int nW = N3H * D_IN;

    // 1. convert x and (transposed) weights to bf16
    cvt_f32_bf16<<<(nX + 255) / 256, 256, 0, stream>>>(x, x_bf, nX);
    cvt_transpose<<<(nW + 255) / 256, 256, 0, stream>>>(w0f, wt0f, D_IN, N3H);
    cvt_transpose<<<(nW + 255) / 256, 256, 0, stream>>>(w0b, wt0b, D_IN, N3H);
    cvt_transpose<<<(nW + 255) / 256, 256, 0, stream>>>(w1f, wt1f, D_IN, N3H);
    cvt_transpose<<<(nW + 255) / 256, 256, 0, stream>>>(w1b, wt1b, D_IN, N3H);

    dim3 gemmGrid(N3H / 128, T / 128, 2);

    // 2. layer-0 GEMMs (fwd + bwd in one launch via grid.z)
    wmma_gemm_bf16<<<gemmGrid, 256, 0, stream>>>(x_bf, wt0f, wt0b, Uf, Ub, T, N3H, D_IN);

    // 3. layer-0 scans (highway = original x)
    sru_scan<<<2, 512, 0, stream>>>(Uf, Ub, v0f, b0f, v0b, b0b, x, h1, T);

    // 4. h1 -> bf16
    cvt_f32_bf16<<<(nX + 255) / 256, 256, 0, stream>>>(h1, h1bf, nX);

    // 5. layer-1 GEMMs
    wmma_gemm_bf16<<<gemmGrid, 256, 0, stream>>>(h1bf, wt1f, wt1b, Uf, Ub, T, N3H, D_IN);

    // 6. layer-1 scans (highway = h1)
    sru_scan<<<2, 512, 0, stream>>>(Uf, Ub, v1f, b1f, v1b, b1b, h1, h2, T);

    // 7. final FC
    fc_kernel<<<(T + 7) / 8, 256, 0, stream>>>(h2, fc_w, fc_b, outp, T);
}
